// DIN_24094766530911
// MI455X (gfx1250) — matrix operand order repaired and checked by reading, not compiled, not run on hardware
//
#include <hip/hip_runtime.h>

// ---------------- CDNA5 (gfx1250) types ----------------
typedef __attribute__((ext_vector_type(16))) __bf16 v16bf;
typedef __attribute__((ext_vector_type(8)))  __bf16 bf16x8;
typedef __attribute__((ext_vector_type(8)))  float  v8f;

// ---------------- problem constants ----------------
#define BB 2048
#define TT 200
#define KD 128     // keys dim
#define D1 80
#define D2 40
#define DJ 384     // D_JOIN
#define MT 13      // ceil(208/16) M-tiles over T (padded to 208)
#define MASKV (-4294967295.0f)
#define INV_SQRTK 0.08838834764831843f  // 1/sqrt(128)

// ---------------- workspace layout (bytes) ----------------
static constexpr size_t OFF_W1F   = 0;         // bf16 W1 frags: 16kc*5nt*32lane*16e = 81920 B
static constexpr size_t OFF_W2F   = 81920;     // bf16 W2 frags: 3*3*32*16 = 9216 B
static constexpr size_t OFF_ITEMJ = 91136;     // f32 [B][128]  = 1048576
static constexpr size_t OFF_BIAS  = 1139712;   // f32 [B]       = 8192
static constexpr size_t OFF_HATT  = 1147904;   // f32 [B][128]  = 1048576
static constexpr size_t OFF_JOIN  = 2196480;   // f32 [B][384]  = 3145728
static constexpr size_t OFF_MEAN  = 5342208;   // f32 [384]
static constexpr size_t OFF_RSTD  = 5343744;   // f32 [384]
static constexpr size_t OFF_Y1    = 5345280;   // f32 [B][80]   = 655360
static constexpr size_t OFF_Y2    = 6000640;   // f32 [B][40]   = 327680

__device__ __forceinline__ float sigf(float x) { return 1.0f / (1.0f + __expf(-x)); }

__device__ __forceinline__ v16bf pack16(bf16x8 lo, bf16x8 hi) {
  v16bf r;
#pragma unroll
  for (int e = 0; e < 8; ++e) { r[e] = lo[e]; r[e + 8] = hi[e]; }
  return r;
}

// ============ prep: W1 -> bf16 WMMA B-fragment layout ============
// frag element index = ((kc*5+nt)*32 + lane)*16 + e ; K = kc*32 + (lane>>4)*8 + (e&7) + (e>>3)*16 ; N = nt*16 + (lane&15)
__global__ void din_prep_w1(const float* __restrict__ W1, __bf16* __restrict__ w1f) {
  int idx = blockIdx.x * 256 + threadIdx.x;
  if (idx >= 16 * 5 * 32 * 16) return;
  int e = idx & 15, lane = (idx >> 4) & 31, r = idx >> 9;
  int nt = r % 5, kc = r / 5;
  int k = kc * 32 + (lane >> 4) * 8 + (e & 7) + (e >> 3) * 16;
  int n = nt * 16 + (lane & 15);
  w1f[idx] = (__bf16)W1[k * D1 + n];
}

// W2 padded to K=96, N=48 with zeros
__global__ void din_prep_w2(const float* __restrict__ W2, __bf16* __restrict__ w2f) {
  int idx = blockIdx.x * 256 + threadIdx.x;
  if (idx >= 3 * 3 * 32 * 16) return;
  int e = idx & 15, lane = (idx >> 4) & 31, r = idx >> 9;
  int nt = r % 3, kc = r / 3;
  int k = kc * 32 + (lane >> 4) * 8 + (e & 7) + (e >> 3) * 16;
  int n = nt * 16 + (lane & 15);
  float v = (k < D1 && n < D2) ? W2[k * D2 + n] : 0.f;
  w2f[idx] = (__bf16)v;
}

// item_join[b] = concat(item_table[item], cate_table[cate_list[item]]); bias[b]
__global__ void din_prep_item(const int* __restrict__ item, const int* __restrict__ cate_list,
                              const float* __restrict__ item_table, const float* __restrict__ cate_table,
                              const float* __restrict__ item_bias,
                              float* __restrict__ itemj, float* __restrict__ biasb) {
  int b = blockIdx.x, k = threadIdx.x;
  int it = item[b];
  float v = (k < 64) ? item_table[(size_t)it * 64 + k]
                     : cate_table[(size_t)cate_list[it] * 64 + (k - 64)];
  itemj[(size_t)b * KD + k] = v;
  if (k == 0) biasb[b] = item_bias[it];
}

// ============ fused DIN attention: din_all build + 2 WMMA GEMMs + softmax + pooling ============
// LDS layout (bytes)
#define SO_W1  0        // 81920  bf16 W1 frags
#define SO_W2  81920    // 9216   bf16 W2 frags
#define SO_Q   91136    // 256    bf16 q[128]
#define SO_C   91392    // 656    f32: b1[0..79] b2[80..119] W3[120..159] b3[160]
#define SO_HI  92048    // 832    int hist idx[208]
#define SO_CI  92880    // 832    int cate idx[208]
#define SO_S   93712    // 832    f32 scores[208]
#define SO_RED 94544    // 1024   f32 red[256]
#define SO_ACC 95568    // 1024   f32 acc[2][128]
#define SO_SCR 96592    // 8*4096 per-wave scratch (hist tile -> h1 -> h2)
#define SMEM_TOTAL (96592 + 8 * 4096)

__global__ __launch_bounds__(256) void din_attention(
    const int* __restrict__ history, const int* __restrict__ length,
    const int* __restrict__ cate_list,
    const float* __restrict__ item_table, const float* __restrict__ cate_table,
    const float* __restrict__ itemj,
    const __bf16* __restrict__ w1f, const __bf16* __restrict__ w2f,
    const float* __restrict__ att_b1, const float* __restrict__ att_b2,
    const float* __restrict__ att_W3, const float* __restrict__ att_b3,
    float* __restrict__ hist_attn) {
  __shared__ alignas(128) char smem[SMEM_TOTAL];
  __bf16* sW1 = (__bf16*)(smem + SO_W1);
  __bf16* sW2 = (__bf16*)(smem + SO_W2);
  __bf16* sQ  = (__bf16*)(smem + SO_Q);
  float*  sC  = (float*)(smem + SO_C);
  int*    sHI = (int*)(smem + SO_HI);
  int*    sCI = (int*)(smem + SO_CI);
  float*  sS  = (float*)(smem + SO_S);
  float*  sRed = (float*)(smem + SO_RED);
  float*  sAcc = (float*)(smem + SO_ACC);

  int tid = threadIdx.x;
  int b = blockIdx.x;

  // cooperative stage of weight fragments (uint4 = 16B copies)
  {
    const uint4* src = (const uint4*)w1f;
    uint4* dst = (uint4*)sW1;
    for (int i = tid; i < 81920 / 16; i += 256) dst[i] = src[i];
    const uint4* s2 = (const uint4*)w2f;
    uint4* d2 = (uint4*)sW2;
    for (int i = tid; i < 9216 / 16; i += 256) d2[i] = s2[i];
  }
  if (tid < KD) sQ[tid] = (__bf16)itemj[(size_t)b * KD + tid];
  if (tid < 80) sC[tid] = att_b1[tid];
  else if (tid < 120) sC[tid] = att_b2[tid - 80];
  else if (tid < 160) sC[tid] = att_W3[tid - 120];
  else if (tid == 160) sC[160] = att_b3[0];
  for (int t = tid; t < TT; t += 256) {
    int h = history[(size_t)b * TT + t];
    sHI[t] = h;
    sCI[t] = cate_list[h];
  }
  __syncthreads();

  int lane = tid & 31, wv = tid >> 5;
  int mrow = lane & 15, khalf = lane >> 4;
  __bf16* scrB = (__bf16*)(smem + SO_SCR + wv * 4096);
  float*  scrF = (float*)(smem + SO_SCR + wv * 4096);

  for (int m = wv; m < MT; m += 8) {
    int t0 = m * 16;
    // stage hist tile [16][128] bf16 (zeros for t >= 200)
    for (int i = lane; i < 16 * KD; i += 32) {
      int tl = i >> 7, k = i & 127, t = t0 + tl;
      float v = 0.f;
      if (t < TT)
        v = (k < 64) ? item_table[(size_t)sHI[t] * 64 + k]
                     : cate_table[(size_t)sCI[t] * 64 + (k - 64)];
      scrB[i] = (__bf16)v;
    }
    // ---- GEMM1: [16 x 512] x [512 x 80], bf16 WMMA, f32 accum ----
    // Fully-unrolled section loop (sec resolves at compile time => branch-free
    // A-build), with B-fragments batch-loaded *before* the A-build so DS
    // latency overlaps VALU and the 5 WMMAs issue back-to-back.
    v8f acc[5] = {};
#pragma unroll
    for (int sec = 0; sec < 4; ++sec) {       // 0:q 1:hist 2:q-h 3:q*h
#pragma unroll 2
      for (int kc4 = 0; kc4 < 4; ++kc4) {
        int kc = sec * 4 + kc4;
        int kk0 = kc4 * 32 + khalf * 8;       // first 8-elem run within the 128 section
        // batch-load the 5 B fragments first
        v16bf bfr[5];
#pragma unroll
        for (int nt = 0; nt < 5; ++nt)
          bfr[nt] = *(const v16bf*)&sW1[(size_t)(((kc * 5 + nt) * 32) + lane) * 16];
        // build the A fragment for this K-chunk
        v16bf a;
        if (sec == 0) {
          bf16x8 q0 = *(const bf16x8*)&sQ[kk0];
          bf16x8 q1 = *(const bf16x8*)&sQ[kk0 + 16];
          a = pack16(q0, q1);
        } else if (sec == 1) {
          bf16x8 h0 = *(const bf16x8*)&scrB[mrow * KD + kk0];
          bf16x8 h1 = *(const bf16x8*)&scrB[mrow * KD + kk0 + 16];
          a = pack16(h0, h1);
        } else {
          bf16x8 q0 = *(const bf16x8*)&sQ[kk0];
          bf16x8 q1 = *(const bf16x8*)&sQ[kk0 + 16];
          bf16x8 h0 = *(const bf16x8*)&scrB[mrow * KD + kk0];
          bf16x8 h1 = *(const bf16x8*)&scrB[mrow * KD + kk0 + 16];
#pragma unroll
          for (int e = 0; e < 8; ++e) {
            float fq0 = (float)q0[e], fh0 = (float)h0[e];
            float fq1 = (float)q1[e], fh1 = (float)h1[e];
            a[e]     = (__bf16)((sec == 2) ? (fq0 - fh0) : (fq0 * fh0));
            a[e + 8] = (__bf16)((sec == 2) ? (fq1 - fh1) : (fq1 * fh1));
          }
        }
#pragma unroll
        for (int nt = 0; nt < 5; ++nt)
          acc[nt] = __builtin_amdgcn_wmma_f32_16x16x32_bf16(
              false, a, false, bfr[nt], (short)0, acc[nt], false, false);
      }
    }
    // h1 = sigmoid(acc + b1) -> scratch as bf16 [16][96] (hist tile is dead)
#pragma unroll
    for (int nt = 0; nt < 5; ++nt) {
      int n = nt * 16 + mrow;
      float bn = sC[n];
#pragma unroll
      for (int v = 0; v < 8; ++v) {
        int mm = v + 8 * khalf;
        scrB[mm * 96 + n] = (__bf16)sigf(acc[nt][v] + bn);
      }
    }
    if (lane < 16) {
#pragma unroll
      for (int mm = 0; mm < 16; ++mm) scrB[mm * 96 + 80 + lane] = (__bf16)0.f;
    }
    // ---- GEMM2: [16 x 96] x [96 x 48] ----
    v8f acc2[3] = {};
#pragma unroll
    for (int kc = 0; kc < 3; ++kc) {
      int kk0 = kc * 32 + khalf * 8;
      // batch-load B fragments, then A, then 3 back-to-back WMMAs
      v16bf bfr2[3];
#pragma unroll
      for (int nt = 0; nt < 3; ++nt)
        bfr2[nt] = *(const v16bf*)&sW2[(size_t)(((kc * 3 + nt) * 32) + lane) * 16];
      bf16x8 a0 = *(const bf16x8*)&scrB[mrow * 96 + kk0];
      bf16x8 a1 = *(const bf16x8*)&scrB[mrow * 96 + kk0 + 16];
      v16bf a = pack16(a0, a1);
#pragma unroll
      for (int nt = 0; nt < 3; ++nt)
        acc2[nt] = __builtin_amdgcn_wmma_f32_16x16x32_bf16(
            false, a, false, bfr2[nt], (short)0, acc2[nt], false, false);
    }
    // h2 = sigmoid(acc2 + b2) -> scratch f32 [16][40] (h1 is dead)
#pragma unroll
    for (int nt = 0; nt < 3; ++nt) {
      int n = nt * 16 + mrow;
      if (n < D2) {
        float bn = sC[80 + n];
#pragma unroll
        for (int v = 0; v < 8; ++v) {
          int mm = v + 8 * khalf;
          scrF[mm * D2 + n] = sigf(acc2[nt][v] + bn);
        }
      }
    }
    // scores: t = t0 + lane (lanes 0..15)
    if (lane < 16) {
      int t = t0 + lane;
      if (t < TT) {
        float s = sC[160];
        for (int j = 0; j < D2; ++j) s += scrF[lane * D2 + j] * sC[120 + j];
        sS[t] = s;
      }
    }
  }
  __syncthreads();

  // ---- mask + scale + softmax over T=200 ----
  int len = length[b];
  float lm = -3.4e38f;
  for (int t = tid; t < TT; t += 256) {
    float s = (t < len) ? sS[t] * INV_SQRTK : (MASKV * INV_SQRTK);
    sS[t] = s;
    lm = fmaxf(lm, s);
  }
  sRed[tid] = lm;
  __syncthreads();
  for (int st = 128; st > 0; st >>= 1) {
    if (tid < st) sRed[tid] = fmaxf(sRed[tid], sRed[tid + st]);
    __syncthreads();
  }
  float mx = sRed[0];
  __syncthreads();
  float ls = 0.f;
  for (int t = tid; t < TT; t += 256) {
    float e = __expf(sS[t] - mx);
    sS[t] = e;
    ls += e;
  }
  sRed[tid] = ls;
  __syncthreads();
  for (int st = 128; st > 0; st >>= 1) {
    if (tid < st) sRed[tid] += sRed[tid + st];
    __syncthreads();
  }
  float inv = 1.f / sRed[0];
  __syncthreads();
  for (int t = tid; t < TT; t += 256) sS[t] *= inv;
  __syncthreads();

  // ---- weighted pooling: hist_attn[b] = sum_t attn[t] * hist_join[b,t,:] (f32 gathers) ----
  {
    int k = tid & 127, half = tid >> 7;
    float a = 0.f;
    for (int t = half; t < TT; t += 2) {
      float v = (k < 64) ? item_table[(size_t)sHI[t] * 64 + k]
                         : cate_table[(size_t)sCI[t] * 64 + (k - 64)];
      a += sS[t] * v;
    }
    sAcc[tid] = a;
    __syncthreads();
    if (tid < 128) hist_attn[(size_t)b * KD + tid] = sAcc[tid] + sAcc[tid + 128];
  }
}

// ============ per-column batch statistics (mean, rstd) ============
__global__ void din_col_stats(const float* __restrict__ X, int C,
                              float* __restrict__ mean, float* __restrict__ rstd) {
  int c = blockIdx.x, tid = threadIdx.x;
  float s = 0.f, q = 0.f;
  for (int r = tid; r < BB; r += 256) {
    float v = X[(size_t)r * C + c];
    s += v;
    q += v * v;
  }
  __shared__ float rs[256], rq[256];
  rs[tid] = s; rq[tid] = q;
  __syncthreads();
  for (int st = 128; st > 0; st >>= 1) {
    if (tid < st) { rs[tid] += rs[tid + st]; rq[tid] += rq[tid + st]; }
    __syncthreads();
  }
  if (tid == 0) {
    float m = rs[0] * (1.f / BB);
    float v = rq[0] * (1.f / BB) - m * m;
    mean[c] = m;
    rstd[c] = rsqrtf(v + 1e-3f);
  }
}

// ============ hist BN + hist_fc + join assembly ============
__global__ void din_hist_fc_join(const float* __restrict__ hist_attn,
                                 const float* __restrict__ mean, const float* __restrict__ rstd,
                                 const float* __restrict__ hg, const float* __restrict__ hb,
                                 const float* __restrict__ hW, const float* __restrict__ hbias,
                                 const int* __restrict__ user, const float* __restrict__ user_table,
                                 const float* __restrict__ itemj, float* __restrict__ join) {
  int b = blockIdx.x, j = threadIdx.x;  // 128 threads
  __shared__ float xn[KD];
  float ha = hist_attn[(size_t)b * KD + j];
  xn[j] = (ha - mean[j]) * rstd[j] * hg[j] + hb[j];
  join[(size_t)b * DJ + j]       = user_table[(size_t)user[b] * 128 + j];
  join[(size_t)b * DJ + 128 + j] = itemj[(size_t)b * KD + j];
  __syncthreads();
  float a = hbias[j];
  for (int k = 0; k < KD; ++k) a += xn[k] * hW[k * KD + j];
  join[(size_t)b * DJ + 256 + j] = a;
}

// ============ final BN + fc1 ============
__global__ void din_fc1(const float* __restrict__ join,
                        const float* __restrict__ mean, const float* __restrict__ rstd,
                        const float* __restrict__ fg, const float* __restrict__ fb,
                        const float* __restrict__ W, const float* __restrict__ bias,
                        float* __restrict__ y1) {
  int b = blockIdx.x, tid = threadIdx.x;  // 128 threads
  __shared__ float xb[DJ];
  for (int i = tid; i < DJ; i += 128) {
    float v = join[(size_t)b * DJ + i];
    xb[i] = (v - mean[i]) * rstd[i] * fg[i] + fb[i];
  }
  __syncthreads();
  if (tid < D1) {
    float a = bias[tid];
    for (int k = 0; k < DJ; ++k) a += xb[k] * W[k * D1 + tid];
    y1[(size_t)b * D1 + tid] = a;
  }
}

// ============ dice(y1) + fc2 ============
__global__ void din_dice_fc2(const float* __restrict__ y1,
                             const float* __restrict__ mean, const float* __restrict__ rstd,
                             const float* __restrict__ alpha, const float* __restrict__ beta,
                             const float* __restrict__ W, const float* __restrict__ bias,
                             float* __restrict__ y2) {
  int b = blockIdx.x, tid = threadIdx.x;  // 128 threads
  __shared__ float x1[D1];
  if (tid < D1) {
    float y = y1[(size_t)b * D1 + tid];
    float xn = (y - mean[tid]) * rstd[tid];
    float p = sigf(beta[tid] * xn);
    x1[tid] = y * (p + alpha[tid] * (1.f - p));
  }
  __syncthreads();
  if (tid < D2) {
    float a = bias[tid];
    for (int k = 0; k < D1; ++k) a += x1[k] * W[k * D2 + tid];
    y2[(size_t)b * D2 + tid] = a;
  }
}

// ============ dice(y2) + fc3 + bias + sigmoid ============
__global__ void din_dice_fc3_out(const float* __restrict__ y2,
                                 const float* __restrict__ mean, const float* __restrict__ rstd,
                                 const float* __restrict__ alpha, const float* __restrict__ beta,
                                 const float* __restrict__ W3, const float* __restrict__ b3,
                                 const float* __restrict__ biasb, float* __restrict__ out) {
  int b = blockIdx.x * 256 + threadIdx.x;
  if (b >= BB) return;
  float a = b3[0] + biasb[b];
  for (int j = 0; j < D2; ++j) {
    float y = y2[(size_t)b * D2 + j];
    float xn = (y - mean[j]) * rstd[j];
    float p = sigf(beta[j] * xn);
    a += y * (p + alpha[j] * (1.f - p)) * W3[j];
  }
  out[b] = a;
  out[BB + b] = sigf(a);
}

// ==================== host launcher ====================
extern "C" void kernel_launch(void* const* d_in, const int* in_sizes, int n_in,
                              void* d_out, int out_size, void* d_ws, size_t ws_size,
                              hipStream_t stream) {
  (void)in_sizes; (void)n_in; (void)out_size; (void)ws_size;
  const int*   user       = (const int*)d_in[0];
  const int*   item       = (const int*)d_in[1];
  const int*   history    = (const int*)d_in[2];
  const int*   length     = (const int*)d_in[3];
  const int*   cate_list  = (const int*)d_in[4];
  const float* user_table = (const float*)d_in[5];
  const float* item_table = (const float*)d_in[6];
  const float* cate_table = (const float*)d_in[7];
  const float* item_bias  = (const float*)d_in[8];
  const float* att_W1 = (const float*)d_in[9];
  const float* att_b1 = (const float*)d_in[10];
  const float* att_W2 = (const float*)d_in[11];
  const float* att_b2 = (const float*)d_in[12];
  const float* att_W3 = (const float*)d_in[13];
  const float* att_b3 = (const float*)d_in[14];
  const float* hbn_g  = (const float*)d_in[15];
  const float* hbn_b  = (const float*)d_in[16];
  const float* hist_W = (const float*)d_in[17];
  const float* hist_b = (const float*)d_in[18];
  const float* fbn_g  = (const float*)d_in[19];
  const float* fbn_b  = (const float*)d_in[20];
  const float* fc1_W  = (const float*)d_in[21];
  const float* fc1_b  = (const float*)d_in[22];
  const float* d1_a   = (const float*)d_in[23];
  const float* d1_b   = (const float*)d_in[24];
  const float* fc2_W  = (const float*)d_in[25];
  const float* fc2_b  = (const float*)d_in[26];
  const float* d2_a   = (const float*)d_in[27];
  const float* d2_b   = (const float*)d_in[28];
  const float* fc3_W  = (const float*)d_in[29];
  const float* fc3_b  = (const float*)d_in[30];

  char* ws = (char*)d_ws;
  __bf16* w1f  = (__bf16*)(ws + OFF_W1F);
  __bf16* w2f  = (__bf16*)(ws + OFF_W2F);
  float* itemj = (float*)(ws + OFF_ITEMJ);
  float* biasb = (float*)(ws + OFF_BIAS);
  float* hatt  = (float*)(ws + OFF_HATT);
  float* join  = (float*)(ws + OFF_JOIN);
  float* mean  = (float*)(ws + OFF_MEAN);
  float* rstd  = (float*)(ws + OFF_RSTD);
  float* y1    = (float*)(ws + OFF_Y1);
  float* y2    = (float*)(ws + OFF_Y2);
  float* out   = (float*)d_out;

  din_prep_w1<<<160, 256, 0, stream>>>(att_W1, w1f);
  din_prep_w2<<<18, 256, 0, stream>>>(att_W2, w2f);
  din_prep_item<<<BB, 128, 0, stream>>>(item, cate_list, item_table, cate_table,
                                        item_bias, itemj, biasb);

  din_attention<<<BB, 256, 0, stream>>>(history, length, cate_list, item_table, cate_table,
                                        itemj, w1f, w2f, att_b1, att_b2, att_W3, att_b3, hatt);

  din_col_stats<<<KD, 256, 0, stream>>>(hatt, KD, mean, rstd);
  din_hist_fc_join<<<BB, 128, 0, stream>>>(hatt, mean, rstd, hbn_g, hbn_b, hist_W, hist_b,
                                           user, user_table, itemj, join);
  din_col_stats<<<DJ, 256, 0, stream>>>(join, DJ, mean, rstd);
  din_fc1<<<BB, 128, 0, stream>>>(join, mean, rstd, fbn_g, fbn_b, fc1_W, fc1_b, y1);
  din_col_stats<<<D1, 256, 0, stream>>>(y1, D1, mean, rstd);
  din_dice_fc2<<<BB, 128, 0, stream>>>(y1, mean, rstd, d1_a, d1_b, fc2_W, fc2_b, y2);
  din_col_stats<<<D2, 256, 0, stream>>>(y2, D2, mean, rstd);
  din_dice_fc3_out<<<8, 256, 0, stream>>>(y2, mean, rstd, d2_a, d2_b, fc3_W, fc3_b, biasb, out);
}
